// TIRGRNN_71081708749028
// MI455X (gfx1250) — matrix operand order repaired and checked by reading, not compiled, not run on hardware
//
#include <hip/hip_runtime.h>
#include <hip/hip_bf16.h>

// Problem constants (match reference)
constexpr int B = 128;
constexpr int T = 128;
constexpr int H = 1024;
constexpr int L = 2;
constexpr int N3H = 3 * H;        // 3072
constexpr int MROWS = B * T;      // 16384
constexpr int KSTEP = 32;         // K per WMMA
constexpr int SSTR = 40;          // padded LDS row stride (elements) to spread banks

typedef __bf16 bf16;
typedef __attribute__((ext_vector_type(16))) __bf16 v16bf;
typedef __attribute__((ext_vector_type(8)))  __bf16 v8bf;
typedef __attribute__((ext_vector_type(4)))  __bf16 v4bf;
typedef __attribute__((ext_vector_type(8)))  float  v8f;
typedef __attribute__((ext_vector_type(4)))  int    v4i;
typedef __attribute__((ext_vector_type(2)))  int    v2i;

#define GLOBAL_AS __attribute__((address_space(1)))
#define LDS_AS    __attribute__((address_space(3)))

// Async global->LDS path (CDNA5 GLOBAL_LOAD_ASYNC_TO_LDS_*, ASYNCcnt).
// Guarded: if the builtins are absent we fall back to plain LDS staging.
#if defined(__AMDGCN__) &&                                                   \
    __has_builtin(__builtin_amdgcn_global_load_async_to_lds_b128) &&         \
    __has_builtin(__builtin_amdgcn_global_load_async_to_lds_b64) &&          \
    __has_builtin(__builtin_amdgcn_s_wait_asynccnt)
#define USE_ASYNC_LDS 1
#else
#define USE_ASYNC_LDS 0
#endif

__device__ __forceinline__ void stage_b128(const bf16* g, bf16* l) {
#if USE_ASYNC_LDS
    __builtin_amdgcn_global_load_async_to_lds_b128(
        (GLOBAL_AS v4i*)g, (LDS_AS v4i*)l, 0, 0);
#else
    *(v8bf*)l = *(const v8bf*)g;
#endif
}

__device__ __forceinline__ void stage_b64(const bf16* g, bf16* l) {
#if USE_ASYNC_LDS
    __builtin_amdgcn_global_load_async_to_lds_b64(
        (GLOBAL_AS v2i*)g, (LDS_AS v2i*)l, 0, 0);
#else
    *(v4bf*)l = *(const v4bf*)g;
#endif
}

__device__ __forceinline__ void wait_async_le1() {
#if USE_ASYNC_LDS
    __builtin_amdgcn_s_wait_asynccnt(1);
#endif
}
__device__ __forceinline__ void wait_async_le0() {
#if USE_ASYNC_LDS
    __builtin_amdgcn_s_wait_asynccnt(0);
#endif
}

// ---------------------------------------------------------------------------
// WMMA fragment loaders (v_wmma_f32_16x16x32_bf16 layouts as used here; A/B k order is contiguous, not the ISA table order)
//
// A (16x32, M x K): lanes 0-15 hold M=0..15 with K=0..15 contiguous;
//                   lanes 16-31 hold K=16..31.
// B (32x16, K x N): lane n (0-15) holds column n, K=0..15 contiguous;
//                   lanes 16-31 hold the same columns, K=16..31.
// C/D (16x16 f32): VGPR v -> row (v + 8*(lane>=16)), col = lane&15.
// ---------------------------------------------------------------------------
__device__ __forceinline__ v16bf load_a_frag_g(const bf16* __restrict__ base,
                                               int ld, int m0, int k0, int lane) {
    const bf16* p = base + (size_t)(m0 + (lane & 15)) * ld + k0 + ((lane >> 4) << 4);
    union { v16bf v; v8bf h[2]; } f;
    f.h[0] = *(const v8bf*)(p);
    f.h[1] = *(const v8bf*)(p + 8);
    return f.v;
}

__device__ __forceinline__ v16bf load_a_frag_lds(const bf16* sbuf, int lane) {
    const bf16* p = sbuf + (lane & 15) * SSTR + ((lane >> 4) << 4);
    union { v16bf v; v8bf h[2]; } f;
    f.h[0] = *(const v8bf*)(p);
    f.h[1] = *(const v8bf*)(p + 8);
    return f.v;
}

__device__ __forceinline__ v16bf load_b_frag_g(const bf16* __restrict__ base,
                                               int ld, int n0, int k0, int lane) {
    const bf16* p = base + (size_t)(n0 + (lane & 15)) * ld + k0 + ((lane >> 4) << 4);
    union { v16bf v; v8bf h[2]; } f;
    f.h[0] = *(const v8bf*)(p);
    f.h[1] = *(const v8bf*)(p + 8);
    return f.v;
}

__device__ __forceinline__ v16bf load_b_frag_lds(const bf16* sbuf, int n_local, int lane) {
    const bf16* p = sbuf + (n_local + (lane & 15)) * SSTR + ((lane >> 4) << 4);
    union { v16bf v; v8bf h[2]; } f;
    f.h[0] = *(const v8bf*)(p);
    f.h[1] = *(const v8bf*)(p + 8);
    return f.v;
}

// ---------------------------------------------------------------------------
// f32 -> bf16 conversion
// ---------------------------------------------------------------------------
__global__ void cvt_kernel(const float* __restrict__ in, bf16* __restrict__ out, int n) {
    int i = blockIdx.x * blockDim.x + threadIdx.x;
    if (i < n) out[i] = (bf16)in[i];
}

__global__ void init_h_kernel(const float* __restrict__ src,
                              float* __restrict__ hf, bf16* __restrict__ hb, int n) {
    int i = blockIdx.x * blockDim.x + threadIdx.x;
    if (i < n) {
        float v = src[i];
        hf[i] = v;
        hb[i] = (bf16)v;
    }
}

// ---------------------------------------------------------------------------
// Input projection GEMM: xp[r, n] = sum_k cur[r,k] * Wx[n,k] + bx[n]
// Grid: (MROWS/128, 3H/64), block 256 (8 waves). Wave w owns a 16x64 tile.
// The 64x32 B-tile (shared by all 8 waves) is staged in LDS via async DMA,
// double-buffered; A fragments are per-wave and stay on the direct path.
// ---------------------------------------------------------------------------
__global__ void xproj_kernel(const bf16* __restrict__ A,
                             const bf16* __restrict__ W,
                             const float* __restrict__ bias,
                             float* __restrict__ xp) {
    __shared__ __align__(16) bf16 sB[2][64 * SSTR];

    const int tid  = threadIdx.x;
    const int lane = tid & 31;
    const int wave = tid >> 5;
    const int m0 = blockIdx.x * 128 + wave * 16;
    const int n0 = blockIdx.y * 64;

    // staging map: 256 threads x 16B = 64 rows x 64B
    const int srow = tid >> 2;            // 0..63
    const int schk = (tid & 3) * 8;       // element offset in row
    const bf16* gB = W + (size_t)(n0 + srow) * H + schk;
    bf16* lB[2] = { &sB[0][srow * SSTR + schk], &sB[1][srow * SSTR + schk] };

    v8f acc[4] = {};

    stage_b128(gB, lB[0]);                // prologue: k=0 -> buf0

    for (int k = 0; k < H; k += KSTEP) {
        const int cur = (k >> 5) & 1;
        if (k + KSTEP < H) {
            stage_b128(gB + k + KSTEP, lB[cur ^ 1]);
            wait_async_le1();
        } else {
            wait_async_le0();
        }
        __syncthreads();

        __builtin_prefetch(A + (size_t)(m0 + (lane & 15)) * H + k + KSTEP, 0, 1);
        v16bf a = load_a_frag_g(A, H, m0, k, lane);
        const bf16* sbuf = sB[cur];
#pragma unroll
        for (int j = 0; j < 4; ++j) {
            v16bf b = load_b_frag_lds(sbuf, j * 16, lane);
            acc[j] = __builtin_amdgcn_wmma_f32_16x16x32_bf16(
                false, a, false, b, (short)0, acc[j], false, false);
        }
        __syncthreads();
    }

#pragma unroll
    for (int j = 0; j < 4; ++j) {
        const int col = n0 + j * 16 + (lane & 15);
        const float bv = bias[col];
#pragma unroll
        for (int v = 0; v < 8; ++v) {
            const int row = m0 + v + ((lane >> 4) << 3);
            xp[(size_t)row * N3H + col] = acc[j][v] + bv;
        }
    }
}

// ---------------------------------------------------------------------------
// One GRU time step. Grid: (B/16, H/64), block 128 (4 waves).
// The 16x32 hidden-state A-tile (shared by all 4 waves) is staged in LDS via
// async DMA, double-buffered. Per-wave Wh B-fragments stream from L2 directly.
// ---------------------------------------------------------------------------
__global__ void gru_step_kernel(const bf16*  __restrict__ h_in_b,
                                const float* __restrict__ h_in_f,
                                bf16*        __restrict__ h_out_b,
                                float*       __restrict__ h_out_f,
                                const bf16*  __restrict__ wh,    // [3H x H]
                                const float* __restrict__ bh,    // [3H]
                                const float* __restrict__ xp,    // [MROWS x 3H]
                                bf16*        __restrict__ next_cur, // layer0: [B,T,H] bf16
                                float*       __restrict__ seq_out,  // layer1: d_out [B,T,H]
                                float*       __restrict__ last_out, // d_out tail slice [B,H]
                                int t) {
    __shared__ __align__(16) bf16 sA[2][16 * SSTR];

    const int tid  = threadIdx.x;
    const int lane = tid & 31;
    const int wave = tid >> 5;
    const int m0 = blockIdx.x * 16;                 // batch tile
    const int o0 = blockIdx.y * 64 + wave * 16;     // H (output-column) tile

    // staging map: 128 threads x 8B = 16 rows x 64B
    const int srow = tid >> 3;            // 0..15
    const int schk = (tid & 7) * 4;       // element offset in row
    const bf16* gA = h_in_b + (size_t)(m0 + srow) * H + schk;
    bf16* lA[2] = { &sA[0][srow * SSTR + schk], &sA[1][srow * SSTR + schk] };

    v8f ar = {}, az = {}, an = {};

    stage_b64(gA, lA[0]);                 // prologue: k=0 -> buf0

    for (int k = 0; k < H; k += KSTEP) {
        const int cur = (k >> 5) & 1;
        if (k + KSTEP < H) {
            stage_b64(gA + k + KSTEP, lA[cur ^ 1]);
            wait_async_le1();
        } else {
            wait_async_le0();
        }
        __syncthreads();

        v16bf a  = load_a_frag_lds(sA[cur], lane);
        v16bf br = load_b_frag_g(wh, H, 0 * H + o0, k, lane);
        v16bf bz = load_b_frag_g(wh, H, 1 * H + o0, k, lane);
        v16bf bn = load_b_frag_g(wh, H, 2 * H + o0, k, lane);
        ar = __builtin_amdgcn_wmma_f32_16x16x32_bf16(false, a, false, br, (short)0, ar, false, false);
        az = __builtin_amdgcn_wmma_f32_16x16x32_bf16(false, a, false, bz, (short)0, az, false, false);
        an = __builtin_amdgcn_wmma_f32_16x16x32_bf16(false, a, false, bn, (short)0, an, false, false);
        __syncthreads();
    }

    const int o = o0 + (lane & 15);
    const float bhr = bh[0 * H + o];
    const float bhz = bh[1 * H + o];
    const float bhn = bh[2 * H + o];

#pragma unroll
    for (int v = 0; v < 8; ++v) {
        const int b = m0 + v + ((lane >> 4) << 3);
        const size_t xrow = ((size_t)b * T + t) * N3H;
        const float xr = xp[xrow + 0 * H + o];
        const float xz = xp[xrow + 1 * H + o];
        const float xn = xp[xrow + 2 * H + o];

        const float r  = 1.0f / (1.0f + __expf(-(xr + ar[v] + bhr)));
        const float z  = 1.0f / (1.0f + __expf(-(xz + az[v] + bhz)));
        const float nn = tanhf(xn + r * (an[v] + bhn));
        const float hold = h_in_f[(size_t)b * H + o];
        const float hnew = (1.0f - z) * nn + z * hold;

        h_out_f[(size_t)b * H + o] = hnew;
        h_out_b[(size_t)b * H + o] = (bf16)hnew;
        if (next_cur) next_cur[((size_t)b * T + t) * H + o] = (bf16)hnew;
        if (seq_out)  seq_out[((size_t)b * T + t) * H + o] = hnew;
        if (t == T - 1) last_out[(size_t)b * H + o] = hnew;
    }
}

// ---------------------------------------------------------------------------
// Host-side orchestration
// ---------------------------------------------------------------------------
extern "C" void kernel_launch(void* const* d_in, const int* in_sizes, int n_in,
                              void* d_out, int out_size, void* d_ws, size_t ws_size,
                              hipStream_t stream) {
    const float* input = (const float*)d_in[0];   // [B,T,H]
    const float* prevh = (const float*)d_in[1];   // [L,B,H]
    const float* Wx    = (const float*)d_in[2];   // [L,3,H,H]
    const float* Wh    = (const float*)d_in[3];   // [L,3,H,H]
    const float* bx    = (const float*)d_in[4];   // [L,3,H]
    const float* bh    = (const float*)d_in[5];   // [L,3,H]
    float* out = (float*)d_out;                   // [B,T,H] ++ [L,B,H]
    (void)in_sizes; (void)n_in; (void)out_size; (void)ws_size;

    char* ws = (char*)d_ws;
    size_t off = 0;
    auto alloc = [&](size_t bytes) -> char* {
        char* p = ws + off;
        off += (bytes + 255) & ~(size_t)255;
        return p;
    };

    bf16*  cur0 = (bf16*)alloc((size_t)MROWS * H * sizeof(bf16));
    bf16*  cur1 = (bf16*)alloc((size_t)MROWS * H * sizeof(bf16));
    bf16*  wxb  = (bf16*)alloc((size_t)L * N3H * H * sizeof(bf16));
    bf16*  whb  = (bf16*)alloc((size_t)L * N3H * H * sizeof(bf16));
    float* xp   = (float*)alloc((size_t)MROWS * N3H * sizeof(float));
    float* hf   = (float*)alloc((size_t)2 * B * H * sizeof(float));
    bf16*  hb   = (bf16*)alloc((size_t)2 * B * H * sizeof(bf16));

    {
        int n = MROWS * H;
        cvt_kernel<<<(n + 255) / 256, 256, 0, stream>>>(input, cur0, n);
        int nw = L * N3H * H;
        cvt_kernel<<<(nw + 255) / 256, 256, 0, stream>>>(Wx, wxb, nw);
        cvt_kernel<<<(nw + 255) / 256, 256, 0, stream>>>(Wh, whb, nw);
    }

    for (int l = 0; l < L; ++l) {
        const bf16*  cur  = (l == 0) ? cur0 : cur1;
        const bf16*  wxl  = wxb + (size_t)l * N3H * H;
        const bf16*  whl  = whb + (size_t)l * N3H * H;
        const float* bxl  = bx + (size_t)l * N3H;
        const float* bhl  = bh + (size_t)l * N3H;

        {
            int n = B * H;
            init_h_kernel<<<(n + 255) / 256, 256, 0, stream>>>(prevh + (size_t)l * B * H, hf, hb, n);
        }

        xproj_kernel<<<dim3(MROWS / 128, N3H / 64), 256, 0, stream>>>(cur, wxl, bxl, xp);

        for (int t = 0; t < T; ++t) {
            const int bi = t & 1;
            const int bo = bi ^ 1;
            gru_step_kernel<<<dim3(B / 16, H / 64), 128, 0, stream>>>(
                hb + (size_t)bi * B * H, hf + (size_t)bi * B * H,
                hb + (size_t)bo * B * H, hf + (size_t)bo * B * H,
                whl, bhl, xp,
                (l == 0) ? cur1 : nullptr,
                (l == L - 1) ? out : nullptr,
                out + (size_t)B * T * H + (size_t)l * B * H,
                t);
        }
    }
}
